// ConvGRU_89223650607385
// MI455X (gfx1250) — matrix operand order repaired and checked by reading, not compiled, not run on hardware
//
#include <hip/hip_runtime.h>
#include <hip/hip_bf16.h>
#include <math.h>

// ---------------------------------------------------------------------------
// MI455X ConvGRU forward.
// Heavy 5x5 GRU convs -> implicit GEMM on v_wmma_f32_16x16x32_f16 (f16 in,
// f32 accumulate).  K padded to 32 slots per input channel (25 taps used) so
// one K-chunk == one channel and pixel math is shifts.  Chunk-invariant
// im2col offsets are precomputed into registers; LDS is double-buffered and
// the K-loop is software-pipelined (stage c+1 while WMMAs consume c) with a
// single barrier per chunk.  Epilogues fuse sigmoid/r*h (zr) and tanh + GRU
// state update (cand).
// ---------------------------------------------------------------------------

typedef __attribute__((ext_vector_type(16))) _Float16 v16h;
typedef __attribute__((ext_vector_type(8)))  _Float16 h8;
typedef __attribute__((ext_vector_type(8)))  float    v8f;

union V16U { v16h v; h8 h[2]; };

__device__ __forceinline__ float sigmoidf_(float x) { return 1.f / (1.f + __expf(-x)); }
__device__ __forceinline__ float lreluf_(float x)   { return x >= 0.f ? x : 0.2f * x; }

// LDS row stride (halves): 32 data + 8 pad -> conflict-free b128 frag reads
#define LDSW 40

// ---------------------------------------------------------------------------
// Implicit-GEMM conv, 256 threads = 8 waves.
//   WN==2 (N=128): waves 4(M) x 2(N), M_block = 64
//   WN==1 (N=64) : waves 8(M) x 1(N), M_block = 128
// Output spatial is always 32x32 (Hout*Wout = 1024).
// MODE 0: out_f16 = lrelu(acc+b)                                (enc conv2)
// MODE 1: s=sigmoid(acc+b); n<64 -> rh16=s*h ; n>=64 -> zf32=s  (GRU zr)
// MODE 2: c=tanh(acc+b); h = (1-z)h + z*c -> hf32,h16[,hs16]    (GRU cand)
// ---------------------------------------------------------------------------
template<int WN, int MODE>
__launch_bounds__(256)
__global__ void conv_wmma(
    const _Float16* __restrict__ src0, int s0_stride, int C0,
    const _Float16* __restrict__ src1, int s1_stride, int C1,
    const _Float16* __restrict__ wpk,  // packed f16 [N][(C0+C1)*32]
    const float*    __restrict__ bias,
    int Hin, int Win, int stride, int pad, int ksize,
    _Float16* __restrict__ out_f16,
    float*    __restrict__ hf32,
    float*    __restrict__ zf32,
    _Float16* __restrict__ rh16,
    _Float16* __restrict__ h16,
    _Float16* __restrict__ hs16)
{
    constexpr int WM = 8 / WN;
    constexpr int Mb = WM << 4;            // 128 or 64
    constexpr int N  = WN << 6;            // 64 or 128
    constexpr int EA = (Mb * 32) / 256;    // staged A elems / thread (16 or 8)

    const int Ctot = C0 + C1;
    const int Kpk  = Ctot << 5;            // packed weight row length (halves)
    const int HW   = Hin * Win;
    const int tid  = threadIdx.x;
    const int mbase = blockIdx.x * Mb;

    // A tile rows [0,Mb), B tile rows [Mb, Mb+N); Mb+N == 192 either config.
    __shared__ __align__(32) _Float16 S[2][192 * LDSW];

    // ---- chunk-invariant im2col offsets, kept in VGPRs ----
    const int ki  = tid & 31;              // tap slot (fixed per thread)
    const int mi0 = tid >> 5;
    int dy = ki / ksize;
    int dx = ki - dy * ksize;
    const bool tap_ok = ki < ksize * ksize;

    int p_img[EA], p_pix[EA];
#pragma unroll
    for (int e = 0; e < EA; ++e) {
        int m   = mbase + mi0 + (e << 3);
        int img = m >> 10, pix = m & 1023;
        int oy  = pix >> 5, ox = pix & 31;
        int iy  = oy * stride - pad + dy;
        int ix  = ox * stride - pad + dx;
        bool ok = tap_ok && (unsigned)iy < (unsigned)Hin && (unsigned)ix < (unsigned)Win;
        p_img[e] = img;
        p_pix[e] = ok ? (iy * Win + ix) : -1;
    }

    const int wave = tid >> 5, lane = tid & 31;
    const int wm = wave % WM, wn = wave / WM;
    const int mrow = wm << 4;              // wave's row base in A tile
    const int nb   = wn << 6;              // wave's col base (64 cols)
    const int r    = lane & 15, g = lane >> 4;

    auto stage = [&](int c, int b) {
        const _Float16* sp;
        int cc, sstride;
        if (c < C0) { sp = src0; cc = c;      sstride = s0_stride; }
        else        { sp = src1; cc = c - C0; sstride = s1_stride; }
        const int coff = cc * HW;
#pragma unroll
        for (int e = 0; e < EA; ++e) {
            _Float16 v = (_Float16)0.f;
            if (p_pix[e] >= 0) v = sp[p_img[e] * sstride + coff + p_pix[e]];
            S[b][(mi0 + (e << 3)) * LDSW + ki] = v;
        }
#pragma unroll
        for (int s2 = 0; s2 < WN; ++s2) {
            int s = tid + (s2 << 8);
            int n = s >> 2, q = s & 3;
            *reinterpret_cast<h8*>(&S[b][(Mb + n) * LDSW + (q << 3)]) =
                *reinterpret_cast<const h8*>(&wpk[n * Kpk + (c << 5) + (q << 3)]);
        }
    };

    v8f acc0 = {}, acc1 = {}, acc2 = {}, acc3 = {};

    auto compute = [&](int b) {
        // A 16x32: lane(g,r): halves = K[16g..16g+15] of row r
        V16U A;
        A.h[0] = *reinterpret_cast<const h8*>(&S[b][(mrow + r) * LDSW + (g << 4)]);
        A.h[1] = *reinterpret_cast<const h8*>(&S[b][(mrow + r) * LDSW + (g << 4) + 8]);
        // B 32x16: lane(g,n): halves = K[16g..16g+15] of column n
        const int rb = Mb + nb + r;
        V16U B0, B1, B2, B3;
        B0.h[0] = *reinterpret_cast<const h8*>(&S[b][(rb +  0) * LDSW + (g << 4)]);
        B0.h[1] = *reinterpret_cast<const h8*>(&S[b][(rb +  0) * LDSW + (g << 4) + 8]);
        B1.h[0] = *reinterpret_cast<const h8*>(&S[b][(rb + 16) * LDSW + (g << 4)]);
        B1.h[1] = *reinterpret_cast<const h8*>(&S[b][(rb + 16) * LDSW + (g << 4) + 8]);
        B2.h[0] = *reinterpret_cast<const h8*>(&S[b][(rb + 32) * LDSW + (g << 4)]);
        B2.h[1] = *reinterpret_cast<const h8*>(&S[b][(rb + 32) * LDSW + (g << 4) + 8]);
        B3.h[0] = *reinterpret_cast<const h8*>(&S[b][(rb + 48) * LDSW + (g << 4)]);
        B3.h[1] = *reinterpret_cast<const h8*>(&S[b][(rb + 48) * LDSW + (g << 4) + 8]);
        acc0 = __builtin_amdgcn_wmma_f32_16x16x32_f16(false, A.v, false, B0.v, (short)0, acc0, false, false);
        acc1 = __builtin_amdgcn_wmma_f32_16x16x32_f16(false, A.v, false, B1.v, (short)0, acc1, false, false);
        acc2 = __builtin_amdgcn_wmma_f32_16x16x32_f16(false, A.v, false, B2.v, (short)0, acc2, false, false);
        acc3 = __builtin_amdgcn_wmma_f32_16x16x32_f16(false, A.v, false, B3.v, (short)0, acc3, false, false);
    };

    // ---- software-pipelined K loop: one barrier per chunk ----
    stage(0, 0);
    __syncthreads();
    for (int c = 0; c < Ctot; ++c) {
        int p = c & 1;
        if (c + 1 < Ctot) stage(c + 1, p ^ 1);  // VMEM for next chunk in flight
        compute(p);                              // ds reads + 4x WMMA
        __syncthreads();
    }

    // ---- epilogue: C/D layout lane(g,r): col n = nb+16j+r, row = 8g+rr ----
    auto epi = [&](v8f a, int j) {
        int n = nb + (j << 4) + r;
        float bn = bias[n];
#pragma unroll
        for (int rr = 0; rr < 8; ++rr) {
            int m   = mbase + mrow + (g << 3) + rr;
            int img = m >> 10, pix = m & 1023;
            float v = a[rr] + bn;
            if constexpr (MODE == 0) {
                out_f16[(img * N + n) * 1024 + pix] = (_Float16)lreluf_(v);
            } else if constexpr (MODE == 1) {
                float s = sigmoidf_(v);
                if (n < 64) {
                    int id = ((img << 6) + n) * 1024 + pix;
                    rh16[id] = (_Float16)(s * hf32[id]);
                } else {
                    int id = ((img << 6) + (n - 64)) * 1024 + pix;
                    zf32[id] = s;
                }
            } else {
                float cd = tanhf(v);
                int id = ((img << 6) + n) * 1024 + pix;
                float z  = zf32[id];
                float hp = hf32[id];
                float hn = (1.f - z) * hp + z * cd;
                hf32[id] = hn;
                _Float16 hh = (_Float16)hn;
                h16[id] = hh;
                if (hs16) hs16[id] = hh;
            }
        }
    };
    epi(acc0, 0); epi(acc1, 1); epi(acc2, 2); epi(acc3, 3);
}

// ---------------------------------------------------------------------------
// Pack f32 OIHW weights -> f16 [N][Cuse*32] with tap slot ki = dy*ks+dx,
// zero padded slots; supports a channel-slice offset (decoder [:, CH:]).
// ---------------------------------------------------------------------------
__global__ void pack_w(const float* __restrict__ src, _Float16* __restrict__ dst,
                       int N, int Csrc, int coff, int Cuse, int ksize)
{
    int idx = blockIdx.x * 256 + threadIdx.x;
    if (idx >= N * Cuse * 32) return;
    int ki = idx & 31;
    int t  = idx >> 5;
    int c  = t % Cuse;
    int n  = t / Cuse;
    int kt = ksize * ksize;
    float v = 0.f;
    if (ki < kt) v = src[(n * Csrc + coff + c) * kt + ki];
    dst[(n * Cuse + c) * 32 + ki] = (_Float16)v;
}

// enc conv1: (320,1,128,128) -> (320,16,64,64), 3x3 s2 p1, lrelu, f16 out
__global__ void enc_conv1_k(const float* __restrict__ x, const float* __restrict__ w,
                            const float* __restrict__ b, _Float16* __restrict__ out)
{
    int idx = blockIdx.x * 256 + threadIdx.x;           // 20,971,520 total
    int ox = idx & 63, oy = (idx >> 6) & 63, n = (idx >> 12) & 15, img = idx >> 16;
    float acc = b[n];
    const float* xi = x + img * 16384;
#pragma unroll
    for (int dy = 0; dy < 3; ++dy) {
        int iy = 2 * oy - 1 + dy;
        if ((unsigned)iy >= 128u) continue;
#pragma unroll
        for (int dx = 0; dx < 3; ++dx) {
            int ix = 2 * ox - 1 + dx;
            if ((unsigned)ix >= 128u) continue;
            acc += xi[iy * 128 + ix] * w[n * 9 + dy * 3 + dx];
        }
    }
    out[idx] = (_Float16)lreluf_(acc);
}

__global__ void zero_h_k(float* hf32, _Float16* h16)
{
    int idx = blockIdx.x * 256 + threadIdx.x;
    hf32[idx] = 0.f;
    h16[idx]  = (_Float16)0.f;
}

// dec deconv1: hs (t-major, 10x32x64x32x32 f16) -> y1 (320,32,64,64) f16, lrelu
// lhs_dilation=2, pad=2, 4x4: taps valid where (o+k-2) even, 0<=..<=62
__global__ void deconv1_k(const _Float16* __restrict__ hs, const float* __restrict__ w,
                          const float* __restrict__ b, _Float16* __restrict__ out)
{
    int idx = blockIdx.x * 256 + threadIdx.x;           // 41,943,040 total
    int ox = idx & 63, oy = (idx >> 6) & 63, n = (idx >> 12) & 31, q = idx >> 17;
    const _Float16* hq = hs + (size_t)q * 65536;        // (t*32+b) image, c-major
    float acc = b[n];
#pragma unroll
    for (int ky = 0; ky < 4; ++ky) {
        int py = oy + ky - 2;
        if (py < 0 || py > 62 || (py & 1)) continue;
        int iy = py >> 1;
#pragma unroll
        for (int kx = 0; kx < 4; ++kx) {
            int px = ox + kx - 2;
            if (px < 0 || px > 62 || (px & 1)) continue;
            int ix = px >> 1;
            const _Float16* hp = hq + iy * 32 + ix;
            const float*    wp = w + n * 1024 + ky * 4 + kx;
            float s = 0.f;
            for (int c = 0; c < 64; ++c)
                s += (float)hp[c * 1024] * wp[c * 16];
            acc += s;
        }
    }
    out[idx] = (_Float16)lreluf_(acc);
}

// dec deconv2: y1 (320,32,64,64) -> d_out (B,T,1,128,128) f32 (no activation)
__global__ void deconv2_k(const _Float16* __restrict__ y1, const float* __restrict__ w,
                          const float* __restrict__ b, float* __restrict__ out)
{
    int idx = blockIdx.x * 256 + threadIdx.x;           // 5,242,880 total
    int ox = idx & 127, oy = (idx >> 7) & 127, q = idx >> 14;
    int t = q >> 5, bb = q & 31;
    const _Float16* yq = y1 + (size_t)q * 131072;       // (q, c, 64, 64)
    float acc = b[0];
#pragma unroll
    for (int ky = 0; ky < 4; ++ky) {
        int py = oy + ky - 2;
        if (py < 0 || py > 126 || (py & 1)) continue;
        int iy = py >> 1;
#pragma unroll
        for (int kx = 0; kx < 4; ++kx) {
            int px = ox + kx - 2;
            if (px < 0 || px > 126 || (px & 1)) continue;
            int ix = px >> 1;
            const _Float16* yp = yq + iy * 64 + ix;
            const float*    wp = w + ky * 4 + kx;
            float s = 0.f;
            for (int c = 0; c < 32; ++c)
                s += (float)yp[c * 4096] * wp[c * 16];
            acc += s;
        }
    }
    out[(bb * 10 + t) * 16384 + oy * 128 + ox] = acc;   // (T,B)->(B,T) transpose
}

// ---------------------------------------------------------------------------
extern "C" void kernel_launch(void* const* d_in, const int* in_sizes, int n_in,
                              void* d_out, int out_size, void* d_ws, size_t ws_size,
                              hipStream_t stream)
{
    (void)in_sizes; (void)n_in; (void)out_size; (void)ws_size;
    const float* x        = (const float*)d_in[0];
    const float* enc_w1   = (const float*)d_in[1];
    const float* enc_b1   = (const float*)d_in[2];
    const float* enc_w2   = (const float*)d_in[3];
    const float* enc_b2   = (const float*)d_in[4];
    const float* egru_wzr = (const float*)d_in[5];
    const float* egru_bzr = (const float*)d_in[6];
    const float* egru_wc  = (const float*)d_in[7];
    const float* egru_bc  = (const float*)d_in[8];
    const float* dgru_wzr = (const float*)d_in[9];
    const float* dgru_bzr = (const float*)d_in[10];
    const float* dgru_wc  = (const float*)d_in[11];
    const float* dgru_bc  = (const float*)d_in[12];
    const float* dec_w1   = (const float*)d_in[13];
    const float* dec_b1   = (const float*)d_in[14];
    const float* dec_w2   = (const float*)d_in[15];
    const float* dec_b2   = (const float*)d_in[16];
    float* out = (float*)d_out;

    char* ws = (char*)d_ws;
    size_t off = 0;
    auto carve = [&](size_t bytes) -> char* {
        char* p = ws + off;
        off = (off + bytes + 255) & ~(size_t)255;
        return p;
    };
    _Float16* e1  = (_Float16*)carve(20971520ull * 2);  // (320,16,64,64)
    _Float16* e2  = (_Float16*)carve(20971520ull * 2);  // (320,64,32,32)
    float*    hf  = (float*)   carve(2097152ull * 4);   // h master f32
    _Float16* h16 = (_Float16*)carve(2097152ull * 2);   // h shadow f16
    float*    zf  = (float*)   carve(2097152ull * 4);   // z gate
    _Float16* rh  = (_Float16*)carve(2097152ull * 2);   // r*h
    _Float16* hs  = (_Float16*)carve(20971520ull * 2);  // decoder states (t-major)
    _Float16* y1  = (_Float16*)carve(41943040ull * 2);  // (320,32,64,64)
    _Float16* pw_e2  = (_Float16*)carve(32768ull * 2);
    _Float16* pw_ezr = (_Float16*)carve(524288ull * 2);
    _Float16* pw_ec  = (_Float16*)carve(262144ull * 2);
    _Float16* pw_dzr = (_Float16*)carve(262144ull * 2);
    _Float16* pw_dc  = (_Float16*)carve(131072ull * 2);

    // ---- weight packing (f32 -> padded f16 GEMM layout) ----
    pack_w<<<128,  256, 0, stream>>>(enc_w2,   pw_e2,  64,  16,  0,  16, 3);
    pack_w<<<2048, 256, 0, stream>>>(egru_wzr, pw_ezr, 128, 128, 0, 128, 5);
    pack_w<<<1024, 256, 0, stream>>>(egru_wc,  pw_ec,  64, 128,  0, 128, 5);
    pack_w<<<1024, 256, 0, stream>>>(dgru_wzr, pw_dzr, 128, 128, 64, 64, 5);
    pack_w<<<512,  256, 0, stream>>>(dgru_wc,  pw_dc,  64, 128, 64, 64, 5);

    // ---- encoder ----
    enc_conv1_k<<<81920, 256, 0, stream>>>(x, enc_w1, enc_b1, e1);
    // enc conv2: Cin=16, 3x3 s2 p1, N=64 -> Mb=128, M=327680
    conv_wmma<1, 0><<<2560, 256, 0, stream>>>(e1, 65536, 16, nullptr, 0, 0,
        pw_e2, enc_b2, 64, 64, 2, 1, 3,
        e2, nullptr, nullptr, nullptr, nullptr, nullptr);

    zero_h_k<<<8192, 256, 0, stream>>>(hf, h16);

    // ---- encoder GRU: xt = e2[:, t] (img stride 10*64*1024) ----
    for (int t = 0; t < 10; ++t) {
        conv_wmma<2, 1><<<512, 256, 0, stream>>>(e2 + (size_t)t * 65536, 655360, 64,
            h16, 65536, 64,
            pw_ezr, egru_bzr, 32, 32, 1, 2, 5,
            nullptr, hf, zf, rh, nullptr, nullptr);
        conv_wmma<1, 2><<<256, 256, 0, stream>>>(e2 + (size_t)t * 65536, 655360, 64,
            rh, 65536, 64,
            pw_ec, egru_bc, 32, 32, 1, 2, 5,
            nullptr, hf, zf, nullptr, h16, nullptr);
    }

    // ---- decoder GRU (input = h only; weights are channel-sliced) ----
    for (int t = 0; t < 10; ++t) {
        conv_wmma<2, 1><<<512, 256, 0, stream>>>(h16, 65536, 64, nullptr, 0, 0,
            pw_dzr, dgru_bzr, 32, 32, 1, 2, 5,
            nullptr, hf, zf, rh, nullptr, nullptr);
        conv_wmma<1, 2><<<256, 256, 0, stream>>>(rh, 65536, 64, nullptr, 0, 0,
            pw_dc, dgru_bc, 32, 32, 1, 2, 5,
            nullptr, hf, zf, nullptr, h16, hs + (size_t)t * 2097152);
    }

    // ---- decoder deconvs ----
    deconv1_k<<<163840, 256, 0, stream>>>(hs, dec_w1, dec_b1, y1);
    deconv2_k<<<20480,  256, 0, stream>>>(y1, dec_w2, dec_b2, out);
}
